// ModelWithgraph_embedding_modif_73375221285172
// MI455X (gfx1250) — compile-verified, matrix operand order checked by reading, not run on hardware
//
#include <hip/hip_runtime.h>

#define NN   50000
#define EE   800000
#define CIN0 64
#define HD   128
#define ECC  16
#define MLPD 256
#define NCD  32
#define GG   512

typedef __attribute__((ext_vector_type(16))) __bf16 v16bf;
typedef __attribute__((ext_vector_type(8)))  float  v8f;

__device__ __forceinline__ unsigned short f32_to_bf16(float f) {
  unsigned int u = __float_as_uint(f);
  u += 0x7FFFu + ((u >> 16) & 1u);   // round-to-nearest-even
  return (unsigned short)(u >> 16);
}

// ---------------- f32 -> bf16 convert, 4-wide ----------------
__global__ void k_f32_to_bf16_v4(const float4* __restrict__ in, ushort4* __restrict__ out, int n4) {
  int i = blockIdx.x * blockDim.x + threadIdx.x;
  if (i < n4) {
    float4 v = in[i];
    ushort4 o;
    o.x = f32_to_bf16(v.x); o.y = f32_to_bf16(v.y);
    o.z = f32_to_bf16(v.z); o.w = f32_to_bf16(v.w);
    out[i] = o;
  }
}

// ---------------- zero fill, 4-wide ----------------
__global__ void k_zero_v4(float4* __restrict__ p, int n4) {
  int i = blockIdx.x * blockDim.x + threadIdx.x;
  if (i < n4) p[i] = make_float4(0.f, 0.f, 0.f, 0.f);
}

// ---------------- in-degree of real edges ----------------
__global__ void k_deg(const int* __restrict__ ei, int* __restrict__ deg) {
  int i = blockIdx.x * blockDim.x + threadIdx.x;
  if (i < EE) atomicAdd(&deg[ei[EE + i]], 1);
}

// ---------------- exclusive scan of deg (3-phase) ----------------
__global__ void k_scan1(const int* __restrict__ deg, int* __restrict__ offs,
                        int* __restrict__ bsum, int n) {
  __shared__ int s[256];
  int i = blockIdx.x * 256 + threadIdx.x;
  int v = (i < n) ? deg[i] : 0;
  s[threadIdx.x] = v;
  __syncthreads();
  for (int d = 1; d < 256; d <<= 1) {
    int t = (threadIdx.x >= d) ? s[threadIdx.x - d] : 0;
    __syncthreads();
    s[threadIdx.x] += t;
    __syncthreads();
  }
  if (i < n) offs[i] = s[threadIdx.x] - v;   // exclusive
  if (threadIdx.x == 255) bsum[blockIdx.x] = s[255];
}
__global__ void k_scan2(const int* __restrict__ bsum, int* __restrict__ boff, int nb) {
  __shared__ int s[256];
  int v = (threadIdx.x < nb) ? bsum[threadIdx.x] : 0;
  s[threadIdx.x] = v;
  __syncthreads();
  for (int d = 1; d < 256; d <<= 1) {
    int t = (threadIdx.x >= d) ? s[threadIdx.x - d] : 0;
    __syncthreads();
    s[threadIdx.x] += t;
    __syncthreads();
  }
  if (threadIdx.x < nb) boff[threadIdx.x] = s[threadIdx.x] - v;
}
__global__ void k_scan3(int* __restrict__ offs, const int* __restrict__ boff, int n) {
  int i = blockIdx.x * 256 + threadIdx.x;
  if (i < n) offs[i] += boff[blockIdx.x];
}
__global__ void k_copy_int(const int* __restrict__ in, int* __restrict__ out, int n) {
  int i = blockIdx.x * blockDim.x + threadIdx.x;
  if (i < n) out[i] = in[i];
}

// ---------------- bucket edges by dst (counting-sort into CSR) ----------------
__global__ void k_csr(const int* __restrict__ ei, int* __restrict__ cursor,
                      int* __restrict__ csr_src, int* __restrict__ csr_e) {
  int i = blockIdx.x * blockDim.x + threadIdx.x;
  if (i < EE) {
    int dst = ei[EE + i];
    int p = atomicAdd(&cursor[dst], 1);
    csr_src[p] = ei[i];
    csr_e[p] = i;
  }
}

// ---------------- EA1[v] = 1 + sum of incident edge_attr (layer-invariant) ----------------
__global__ void k_aggEA(const int* __restrict__ offs, const int* __restrict__ deg,
                        const int* __restrict__ csr_e, const float* __restrict__ ea,
                        float* __restrict__ EA1f) {
  int idx = blockIdx.x * blockDim.x + threadIdx.x;
  if (idx >= NN * ECC) return;
  int v = idx >> 4, c = idx & 15;
  int st = offs[v], dg = deg[v];
  float s = 1.0f;                              // self-loop ea = ones
  for (int k = 0; k < dg; ++k)
    s += ea[(size_t)csr_e[st + k] * ECC + c];
  EA1f[idx] = s;
}

// ---------------- W (fan x 128) -> bf16 padded to fanP rows ----------------
__global__ void k_make_wpad(const float* __restrict__ W, unsigned short* __restrict__ out,
                            int fan, int fanP) {
  int idx = blockIdx.x * blockDim.x + threadIdx.x;
  if (idx >= fanP * HD) return;
  int k = idx >> 7, c = idx & 127;
  out[idx] = (k < fan) ? f32_to_bf16(W[k * HD + c]) : (unsigned short)0;
}

// ---------------- per-node neighbor aggregation + Acat assembly ----------------
// Acat[v] = [ (deg+1)*x[v] | aggX[v]+x[v] | EA1[v] | 0-pad ]  (bf16, fanP = 2*cin+32)
__global__ __launch_bounds__(256)
void k_aggregate(const float* __restrict__ xin, const int* __restrict__ offs,
                 const int* __restrict__ deg, const int* __restrict__ csr_src,
                 const float* __restrict__ EA1f, unsigned short* __restrict__ Acat,
                 int cin, int fanP, int n)
{
  int w = (blockIdx.x * 256 + threadIdx.x) >> 5;   // one wave per node
  int lane = threadIdx.x & 31;
  if (w >= n) return;
  int st = offs[w], dg = deg[w];
  float dp1 = (float)(dg + 1);
  unsigned short* arow = Acat + (size_t)w * fanP;

  if (cin == HD) {
    int c = lane * 4;
    float4 own = *(const float4*)(xin + (size_t)w * HD + c);
    float ax = own.x, ay = own.y, az = own.z, aw4 = own.w;   // U starts at x[v]
    for (int k = 0; k < dg; ++k) {
      int s = csr_src[st + k];
      if (k + 1 < dg) {
        int s2 = csr_src[st + k + 1];
        __builtin_prefetch(xin + (size_t)s2 * HD + c, 0, 0);
      }
      float4 xv = *(const float4*)(xin + (size_t)s * HD + c);
      ax += xv.x; ay += xv.y; az += xv.z; aw4 += xv.w;
    }
    ushort4 o;
    o.x = f32_to_bf16(dp1 * own.x); o.y = f32_to_bf16(dp1 * own.y);
    o.z = f32_to_bf16(dp1 * own.z); o.w = f32_to_bf16(dp1 * own.w);
    *(ushort4*)(arow + c) = o;
    o.x = f32_to_bf16(ax); o.y = f32_to_bf16(ay);
    o.z = f32_to_bf16(az); o.w = f32_to_bf16(aw4);
    *(ushort4*)(arow + HD + c) = o;
  } else {   // cin == 64
    int c = lane * 2;
    float2 own = *(const float2*)(xin + (size_t)w * CIN0 + c);
    float ax = own.x, ay = own.y;
    for (int k = 0; k < dg; ++k) {
      int s = csr_src[st + k];
      if (k + 1 < dg) {
        int s2 = csr_src[st + k + 1];
        __builtin_prefetch(xin + (size_t)s2 * CIN0 + c, 0, 0);
      }
      float2 xv = *(const float2*)(xin + (size_t)s * CIN0 + c);
      ax += xv.x; ay += xv.y;
    }
    ushort2 o;
    o.x = f32_to_bf16(dp1 * own.x); o.y = f32_to_bf16(dp1 * own.y);
    *(ushort2*)(arow + c) = o;
    o.x = f32_to_bf16(ax); o.y = f32_to_bf16(ay);
    *(ushort2*)(arow + CIN0 + c) = o;
  }
  // EA1 (16 cols) + zero pad (16 cols): lanes 0..31 cover cols 2cin..2cin+31
  int pc = 2 * cin + lane;
  arow[pc] = (lane < ECC) ? f32_to_bf16(EA1f[(size_t)w * ECC + lane]) : (unsigned short)0;
}

// ---------------- WMMA bf16 GEMM: C = A0@B0 (+ A1@B1) + bias, optional relu,
// optional gather on A rows, optional bf16 shadow copy of C.
// A tile staged via CDNA5 async global->LDS copy (ASYNCcnt). ----------------
__global__ __launch_bounds__(128)
void k_gemm_wmma(const unsigned short* __restrict__ A0, int lda0, const int* __restrict__ g0, int K0,
                 const unsigned short* __restrict__ A1, int lda1, const int* __restrict__ g1, int K1,
                 const unsigned short* __restrict__ B0, int ldb0,
                 const unsigned short* __restrict__ B1, int ldb1,
                 const float* __restrict__ bias,
                 float* __restrict__ C, int ldc,
                 unsigned short* __restrict__ Cb, int ldcb,
                 int M, int relu)
{
  __shared__ __align__(16) unsigned short As[32][40];   // [m][k]  (80B rows keep 16B align)
  __shared__ __align__(16) unsigned short Bs[32][40];   // [n][k]  transposed

  const int t    = threadIdx.x;
  const int lane = t & 31;
  const int wave = t >> 5;
  const int wm   = wave >> 1;           // 2x2 waves -> 32x32 block tile
  const int wn   = wave & 1;
  const int m0   = blockIdx.y * 32;
  const int n0   = blockIdx.x * 32;

  v8f acc = {};

  const int arow = t >> 2;              // 0..31
  const int acg  = (t & 3) * 8;         // 0,8,16,24  (bf16 elements)
  const int bk   = t >> 2;
  const int bng  = (t & 3) * 8;

  const unsigned int ldsA = (unsigned int)(unsigned long long)&As[arow][acg];

  const unsigned short* Aarr[2] = {A0, A1};
  const unsigned short* Barr[2] = {B0, B1};
  const int* Garr[2] = {g0, g1};
  const int ldaA[2] = {lda0, lda1};
  const int ldbA[2] = {ldb0, ldb1};
  const int Karr[2] = {K0, K1};

  const int r  = lane & 15;
  const int kh = (lane >> 4) * 8;       // K sub-offset per ISA A/B fragment layout

  for (int s = 0; s < 2; ++s) {
    const unsigned short* A = Aarr[s];
    if (A == nullptr) continue;
    const unsigned short* B = Barr[s];
    const int* g  = Garr[s];
    const int lda = ldaA[s], ldb = ldbA[s], K = Karr[s];

    const int gm   = m0 + arow;
    const int gmc  = (gm < M) ? gm : 0;   // clamp: OOB A rows only feed OOB C rows (never stored)
    const int grow = g ? g[gmc] : gmc;
    const unsigned short* aRow = A + (size_t)grow * lda;

    for (int k0 = 0; k0 < K; k0 += 32) {
      union { uint4 q; unsigned short h[8]; } bv;
      bv.q = *(const uint4*)(B + (size_t)(k0 + bk) * ldb + n0 + bng);

      __syncthreads();                  // previous iteration done reading LDS

      {
        unsigned long long ga = (unsigned long long)(aRow + k0 + acg);
        asm volatile("global_load_async_to_lds_b128 %0, %1, off"
                     :: "v"(ldsA), "v"(ga) : "memory");
      }
#pragma unroll
      for (int i = 0; i < 8; ++i) Bs[bng + i][bk] = bv.h[i];   // transpose store

      asm volatile("s_wait_asynccnt 0" ::: "memory");
      __syncthreads();

      union { uint4 q[2]; v16bf v; } fa, fb;
      fa.q[0] = *(const uint4*)&As[wm * 16 + r][kh];
      fa.q[1] = *(const uint4*)&As[wm * 16 + r][16 + kh];
      fb.q[0] = *(const uint4*)&Bs[wn * 16 + r][kh];
      fb.q[1] = *(const uint4*)&Bs[wn * 16 + r][16 + kh];

      acc = __builtin_amdgcn_wmma_f32_16x16x32_bf16(false, fa.v, false, fb.v,
                                                    (short)0, acc, false, false);
    }
  }

  const int col  = n0 + wn * 16 + r;
  const int khp  = lane >> 4;
  const float bval = bias ? bias[col] : 0.0f;
#pragma unroll
  for (int vv = 0; vv < 8; ++vv) {
    int row = m0 + wm * 16 + khp * 8 + vv;
    if (row < M) {
      float val = acc[vv] + bval;
      if (relu) val = fmaxf(val, 0.0f);
      C[(size_t)row * ldc + col] = val;
      if (Cb) Cb[(size_t)row * ldcb + col] = f32_to_bf16(val);
    }
  }
}

// ---------------- agg_pre + (deg+1)*b, ReLU, BN(eval), ReLU -> xf (f32) + xb (bf16) ----------------
__global__ void k_post(const float4* __restrict__ aggpre, const int* __restrict__ deg,
                       const float4* __restrict__ b,
                       const float4* __restrict__ gamma, const float4* __restrict__ beta,
                       const float4* __restrict__ rm, const float4* __restrict__ rv,
                       float4* __restrict__ xf, ushort4* __restrict__ xb)
{
  int idx = blockIdx.x * blockDim.x + threadIdx.x;
  if (idx >= NN * (HD / 4)) return;
  int v = idx >> 5, cg = idx & 31;
  float d = (float)(deg[v] + 1);
  float4 ag = aggpre[idx];
  float4 bc = b[cg];
  float4 gm = gamma[cg], be = beta[cg], mu = rm[cg], va = rv[cg];
  float4 o;
  float val;
  val = fmaxf(ag.x + d * bc.x, 0.f);
  o.x = fmaxf((val - mu.x) * rsqrtf(va.x + 1e-5f) * gm.x + be.x, 0.f);
  val = fmaxf(ag.y + d * bc.y, 0.f);
  o.y = fmaxf((val - mu.y) * rsqrtf(va.y + 1e-5f) * gm.y + be.y, 0.f);
  val = fmaxf(ag.z + d * bc.z, 0.f);
  o.z = fmaxf((val - mu.z) * rsqrtf(va.z + 1e-5f) * gm.z + be.z, 0.f);
  val = fmaxf(ag.w + d * bc.w, 0.f);
  o.w = fmaxf((val - mu.w) * rsqrtf(va.w + 1e-5f) * gm.w + be.w, 0.f);
  xf[idx] = o;
  ushort4 ob;
  ob.x = f32_to_bf16(o.x); ob.y = f32_to_bf16(o.y);
  ob.z = f32_to_bf16(o.z); ob.w = f32_to_bf16(o.w);
  xb[idx] = ob;
}

// ---------------- global add pool ----------------
__global__ void k_pool(const float4* __restrict__ xf, const int* __restrict__ batch,
                       float* __restrict__ pooled)
{
  int idx = blockIdx.x * blockDim.x + threadIdx.x;
  if (idx >= NN * (HD / 4)) return;
  int v = idx >> 5, cg = idx & 31;
  float4 xv = xf[idx];
  float* pp = pooled + (size_t)batch[v] * HD + cg * 4;
  atomicAdd(pp + 0, xv.x);
  atomicAdd(pp + 1, xv.y);
  atomicAdd(pp + 2, xv.z);
  atomicAdd(pp + 3, xv.w);
}

// ==================================================================
extern "C" void kernel_launch(void* const* d_in, const int* in_sizes, int n_in,
                              void* d_out, int out_size, void* d_ws, size_t ws_size,
                              hipStream_t stream)
{
  const float* x     = (const float*)d_in[0];
  const int*   ei    = (const int*)d_in[1];
  const float* ea    = (const float*)d_in[2];
  const int*   batch = (const int*)d_in[3];
  const float *Wl[3], *bl[3], *gl[3], *bel[3], *rml[3], *rvl[3];
  for (int i = 0; i < 3; ++i) {
    Wl[i]  = (const float*)d_in[4 + 6 * i + 0];
    bl[i]  = (const float*)d_in[4 + 6 * i + 1];
    gl[i]  = (const float*)d_in[4 + 6 * i + 2];
    bel[i] = (const float*)d_in[4 + 6 * i + 3];
    rml[i] = (const float*)d_in[4 + 6 * i + 4];
    rvl[i] = (const float*)d_in[4 + 6 * i + 5];
  }
  const float* fc1w = (const float*)d_in[22];
  const float* fc1b = (const float*)d_in[23];
  const float* fc2w = (const float*)d_in[24];
  const float* fc2b = (const float*)d_in[25];
  float* out = (float*)d_out;

  char* p = (char*)d_ws;
  auto alloc = [&](size_t bytes) { char* r = p; p += (bytes + 255) & ~(size_t)255; return (void*)r; };

  unsigned short* Acat    = (unsigned short*)alloc((size_t)NN * 288 * 2);
  float*          aggpre  = (float*)alloc((size_t)NN * HD * 4);
  float*          xf      = (float*)alloc((size_t)NN * HD * 4);
  unsigned short* xb      = (unsigned short*)alloc((size_t)NN * HD * 2);
  float*          Hf      = (float*)alloc((size_t)NN * MLPD * 4);
  unsigned short* Hb      = (unsigned short*)alloc((size_t)NN * MLPD * 2);
  float*          pooled  = (float*)alloc((size_t)GG * HD * 4);
  unsigned short* pooledb = (unsigned short*)alloc((size_t)GG * HD * 2);
  unsigned short* Wb      = (unsigned short*)alloc((size_t)288 * HD * 2);
  unsigned short* fc1wb   = (unsigned short*)alloc((size_t)256 * 256 * 2);
  unsigned short* fc2wb   = (unsigned short*)alloc((size_t)256 * 32 * 2);
  int*            deg     = (int*)alloc((size_t)NN * 4);
  int*            offs    = (int*)alloc((size_t)NN * 4);
  int*            cursor  = (int*)alloc((size_t)NN * 4);
  int*            bsum    = (int*)alloc(256 * 4);
  int*            boff    = (int*)alloc(256 * 4);
  int*            csr_src = (int*)alloc((size_t)EE * 4);
  int*            csr_e   = (int*)alloc((size_t)EE * 4);
  float*          EA1f    = (float*)alloc((size_t)NN * ECC * 4);

  const int MT = (NN + 31) / 32;
  const int NB = (NN + 255) / 256;

  // ---- one-time graph structure: deg, CSR, aggregated edge_attr ----
  k_zero_v4<<<(NN / 4 + 255) / 256, 256, 0, stream>>>((float4*)deg, NN / 4);
  k_deg<<<(EE + 255) / 256, 256, 0, stream>>>(ei, deg);
  k_scan1<<<NB, 256, 0, stream>>>(deg, offs, bsum, NN);
  k_scan2<<<1, 256, 0, stream>>>(bsum, boff, NB);
  k_scan3<<<NB, 256, 0, stream>>>(offs, boff, NN);
  k_copy_int<<<NB, 256, 0, stream>>>(offs, cursor, NN);
  k_csr<<<(EE + 255) / 256, 256, 0, stream>>>(ei, cursor, csr_src, csr_e);
  k_aggEA<<<(NN * ECC + 255) / 256, 256, 0, stream>>>(offs, deg, csr_e, ea, EA1f);

  // ---- 3 conv layers ----
  const float* xin = x;
  int cin = CIN0;
  for (int i = 0; i < 3; ++i) {
    int fan  = 2 * cin + ECC;
    int fanP = 2 * cin + 32;
    k_make_wpad<<<(fanP * HD + 255) / 256, 256, 0, stream>>>(Wl[i], Wb, fan, fanP);
    k_aggregate<<<(NN * 32 + 255) / 256, 256, 0, stream>>>(xin, offs, deg, csr_src,
                                                           EA1f, Acat, cin, fanP, NN);
    dim3 gT(HD / 32, MT);
    k_gemm_wmma<<<gT, 128, 0, stream>>>(Acat, fanP, nullptr, fanP,
                                        nullptr, 0, nullptr, 0,
                                        Wb, HD, nullptr, 0,
                                        nullptr, aggpre, HD, nullptr, 0, NN, 0);
    k_post<<<(NN * (HD / 4) + 255) / 256, 256, 0, stream>>>(
        (const float4*)aggpre, deg, (const float4*)bl[i],
        (const float4*)gl[i], (const float4*)bel[i],
        (const float4*)rml[i], (const float4*)rvl[i],
        (float4*)xf, (ushort4*)xb);
    xin = xf;
    cin = HD;
  }

  // ---- head: pool, FC1 (concat via two accumulated GEMM passes), FC2 ----
  k_zero_v4<<<(GG * HD / 4 + 255) / 256, 256, 0, stream>>>((float4*)pooled, GG * HD / 4);
  k_pool<<<(NN * (HD / 4) + 255) / 256, 256, 0, stream>>>((const float4*)xf, batch, pooled);
  k_f32_to_bf16_v4<<<(GG * HD / 4 + 255) / 256, 256, 0, stream>>>(
      (const float4*)pooled, (ushort4*)pooledb, GG * HD / 4);
  k_f32_to_bf16_v4<<<(256 * 256 / 4 + 255) / 256, 256, 0, stream>>>(
      (const float4*)fc1w, (ushort4*)fc1wb, 256 * 256 / 4);
  k_f32_to_bf16_v4<<<(256 * 32 / 4 + 255) / 256, 256, 0, stream>>>(
      (const float4*)fc2w, (ushort4*)fc2wb, 256 * 32 / 4);

  dim3 gF1(MLPD / 32, MT);
  k_gemm_wmma<<<gF1, 128, 0, stream>>>(xb, HD, nullptr, HD,
                                       pooledb, HD, batch, HD,
                                       fc1wb, MLPD, fc1wb + 128 * MLPD, MLPD,
                                       fc1b, Hf, MLPD, Hb, MLPD, NN, 1);

  dim3 gF2(NCD / 32, MT);
  k_gemm_wmma<<<gF2, 128, 0, stream>>>(Hb, MLPD, nullptr, MLPD,
                                       nullptr, 0, nullptr, 0,
                                       fc2wb, NCD, nullptr, 0,
                                       fc2b, out, NCD, nullptr, 0, NN, 0);
}
